// EdgeScoring_15590731285028
// MI455X (gfx1250) — matrix operand order repaired and checked by reading, not compiled, not run on hardware
//
#include <hip/hip_runtime.h>

typedef __attribute__((ext_vector_type(16))) _Float16 v16h;
typedef __attribute__((ext_vector_type(8)))  float    v8f;

#define Bc 8
#define Nc 512
#define Dc 1024
#define Hc 30
#define Hp 32   // H padded to 32 for WMMA K-dim

__device__ __forceinline__ float fast_tanh(float x) {
#if __has_builtin(__builtin_amdgcn_tanhf)
  return __builtin_amdgcn_tanhf(x);
#else
  // tanh(x) = 1 - 2/(exp2(2*log2(e)*x) + 1)  -> v_exp_f32 + v_rcp_f32 (trans pipe)
  float t = __builtin_amdgcn_exp2f(x * 2.8853900817779268f);
  return 1.0f - 2.0f * __builtin_amdgcn_rcpf(t + 1.0f);
#endif
}

// ---------------------------------------------------------------------------
// Stage 1: P = X * [Ua;Wa]^T  via v_wmma_f32_16x16x32_f16
//   X  = concepts reshaped [B*N=4096, D=1024] (f32 -> f16 on the fly)
//   output: up[4096][32] (cols 0..29 valid), wc[4096][32]  (f32, padded w/ 0)
// One wave (32 threads) per 16x16 output tile; grid = (256 Mtiles, 4 Ntiles).
// ---------------------------------------------------------------------------
__global__ __launch_bounds__(32)
void proj_wmma(const float* __restrict__ X, const float* __restrict__ Ua,
               const float* __restrict__ Wa, float* __restrict__ up,
               float* __restrict__ wcb) {
  const int mtile = blockIdx.x;
  const int ntile = blockIdx.y;
  const int lane  = threadIdx.x & 31;
  const int l15   = lane & 15;
  const int hi    = lane >> 4;

  const float* arow = X + (size_t)(mtile * 16 + l15) * Dc;

  // B-matrix column for this lane -> one row of Ua or Wa (contiguous in K!)
  const int c = ntile * 16 + l15;
  const float* wrow;
  bool bvalid;
  float* dst;
  int dcol;
  if (c < Hp) { bvalid = (c < Hc); wrow = Ua + (size_t)c * Dc;        dst = up;  dcol = c; }
  else        { int h = c - Hp; bvalid = (h < Hc); wrow = Wa + (size_t)h * Dc; dst = wcb; dcol = h; }

  v8f acc = {};
  for (int k0 = 0; k0 < Dc; k0 += 32) {
    // A fragment: lanes 0-15 hold K {0..7,16..23}, lanes 16-31 {8..15,24..31}
    const float4* pa = (const float4*)(arow + k0 + hi * 8);
    float4 f0 = pa[0], f1 = pa[1];
    const float4* pb = (const float4*)(arow + k0 + hi * 8 + 16);
    float4 f2 = pb[0], f3 = pb[1];
    v16h a;
    a[0]=(_Float16)f0.x; a[1]=(_Float16)f0.y; a[2]=(_Float16)f0.z; a[3]=(_Float16)f0.w;
    a[4]=(_Float16)f1.x; a[5]=(_Float16)f1.y; a[6]=(_Float16)f1.z; a[7]=(_Float16)f1.w;
    a[8]=(_Float16)f2.x; a[9]=(_Float16)f2.y; a[10]=(_Float16)f2.z; a[11]=(_Float16)f2.w;
    a[12]=(_Float16)f3.x; a[13]=(_Float16)f3.y; a[14]=(_Float16)f3.z; a[15]=(_Float16)f3.w;

    // B fragment: lane holds col l15; lanes 0-15 K {0..7,16..23}, lanes 16-31 {8..15,24..31}
    v16h bm = {};
    if (bvalid) {
      const float4* pw = (const float4*)(wrow + k0 + hi * 8);
      float4 g0 = pw[0], g1 = pw[1], g2 = pw[4], g3 = pw[5];
      bm[0]=(_Float16)g0.x; bm[1]=(_Float16)g0.y; bm[2]=(_Float16)g0.z; bm[3]=(_Float16)g0.w;
      bm[4]=(_Float16)g1.x; bm[5]=(_Float16)g1.y; bm[6]=(_Float16)g1.z; bm[7]=(_Float16)g1.w;
      bm[8]=(_Float16)g2.x; bm[9]=(_Float16)g2.y; bm[10]=(_Float16)g2.z; bm[11]=(_Float16)g2.w;
      bm[12]=(_Float16)g3.x; bm[13]=(_Float16)g3.y; bm[14]=(_Float16)g3.z; bm[15]=(_Float16)g3.w;
    }
    acc = __builtin_amdgcn_wmma_f32_16x16x32_f16(false, a, false, bm, (short)0,
                                                 acc, false, false);
  }
  // D: VGPR r, lane -> (M = r + 8*hi, N = l15)
#pragma unroll
  for (int r = 0; r < 8; ++r) {
    int rowg = mtile * 16 + hi * 8 + r;
    dst[(size_t)rowg * Hp + dcol] = acc[r];
  }
}

// ---------------------------------------------------------------------------
// Stage 2: scores[b,i,j] = sum_h va[h] * tanh(up[b,i,h] + wc[b,j,h])
// Block = 256 thr (8 waves): i-tile of 16 rows x 128 j's. Per wave: 16 j's.
// Iteration t: A = tanh(up[i,:] + wc[j_t,:]) (16x32), B has va in column t,
// C accumulates -> after 16 iters D = 16x16 scores tile (coalesced store).
// ---------------------------------------------------------------------------
__global__ __launch_bounds__(256)
void pair_score_wmma(const float* __restrict__ up, const float* __restrict__ wcb,
                     const float* __restrict__ va, float* __restrict__ out) {
  __shared__ __align__(16) float up_s[16 * 36];    // stride 36 dwords (16B multiple)
  __shared__ __align__(16) float wc_s[128 * 36];

  const int itile = blockIdx.x;
  const int jgrp  = blockIdx.y;
  const int b     = blockIdx.z;
  const int tid   = threadIdx.x;
  const int i0 = itile * 16;
  const int j0 = jgrp * 128;

  // stage up tile (16x32 = 128 float4) and wc tile (128x32 = 1024 float4)
  if (tid < 128) {
    int row = tid >> 3, c4 = (tid & 7) * 4;
    *(float4*)&up_s[row * 36 + c4] =
        *(const float4*)(up + ((size_t)(b * Nc + i0 + row) * Hp + c4));
  }
  for (int v4 = tid; v4 < 1024; v4 += 256) {
    int row = v4 >> 3, c4 = (v4 & 7) * 4;
    *(float4*)&wc_s[row * 36 + c4] =
        *(const float4*)(wcb + ((size_t)(b * Nc + j0 + row) * Hp + c4));
  }
  __syncthreads();

  const int wave = tid >> 5;
  const int lane = tid & 31;
  const int l15  = lane & 15;
  const int hi   = lane >> 4;
  const int hb   = hi * 8;   // A K-runs: [hb..hb+7] and [hb+16..hb+23]

  // up row for this lane (A-matrix row = l15), invariant over j
  float4 u0 = *(const float4*)&up_s[l15 * 36 + hb];
  float4 u1 = *(const float4*)&up_s[l15 * 36 + hb + 4];
  float4 u2 = *(const float4*)&up_s[l15 * 36 + hb + 16];
  float4 u3 = *(const float4*)&up_s[l15 * 36 + hb + 20];

  // va as a B fragment (lane col = l15; lanes 0-15 K {0..7,16..23}, 16-31 {8..15,24..31})
  v16h vaf;
#pragma unroll
  for (int t = 0; t < 16; ++t) {
    int k = (t < 8) ? hb + t : 16 + hb + (t - 8);
    vaf[t] = (k < Hc) ? (_Float16)va[k] : (_Float16)0.0f;
  }
  const v16h vzero = {};

  v8f acc = {};
#pragma unroll
  for (int t = 0; t < 16; ++t) {
    const float* wrow = &wc_s[(wave * 16 + t) * 36];
    float4 w0 = *(const float4*)(wrow + hb);
    float4 w1 = *(const float4*)(wrow + hb + 4);
    float4 w2 = *(const float4*)(wrow + hb + 16);
    float4 w3 = *(const float4*)(wrow + hb + 20);
    v16h a;
    a[0] = (_Float16)fast_tanh(u0.x + w0.x);
    a[1] = (_Float16)fast_tanh(u0.y + w0.y);
    a[2] = (_Float16)fast_tanh(u0.z + w0.z);
    a[3] = (_Float16)fast_tanh(u0.w + w0.w);
    a[4] = (_Float16)fast_tanh(u1.x + w1.x);
    a[5] = (_Float16)fast_tanh(u1.y + w1.y);
    a[6] = (_Float16)fast_tanh(u1.z + w1.z);
    a[7] = (_Float16)fast_tanh(u1.w + w1.w);
    a[8] = (_Float16)fast_tanh(u2.x + w2.x);
    a[9] = (_Float16)fast_tanh(u2.y + w2.y);
    a[10] = (_Float16)fast_tanh(u2.z + w2.z);
    a[11] = (_Float16)fast_tanh(u2.w + w2.w);
    a[12] = (_Float16)fast_tanh(u3.x + w3.x);
    a[13] = (_Float16)fast_tanh(u3.y + w3.y);
    a[14] = (_Float16)fast_tanh(u3.z + w3.z);
    a[15] = (_Float16)fast_tanh(u3.w + w3.w);
    // place va in column t only; accumulate scores for j_t into D column t
    v16h bm = (l15 == t) ? vaf : vzero;
    acc = __builtin_amdgcn_wmma_f32_16x16x32_f16(false, a, false, bm, (short)0,
                                                 acc, false, false);
  }

  // D tile: rows i0 + (r + 8*hi), cols j0 + wave*16 + l15  (coalesced rows)
  const int jb = j0 + wave * 16 + l15;
#pragma unroll
  for (int r = 0; r < 8; ++r) {
    int i = i0 + hi * 8 + r;
    out[((size_t)b * Nc + i) * Nc + jb] = acc[r];
  }
}

extern "C" void kernel_launch(void* const* d_in, const int* in_sizes, int n_in,
                              void* d_out, int out_size, void* d_ws, size_t ws_size,
                              hipStream_t stream) {
  const float* concepts = (const float*)d_in[0];
  const float* Ua = (const float*)d_in[1];
  const float* Wa = (const float*)d_in[2];
  const float* va = (const float*)d_in[3];
  float* out = (float*)d_out;

  float* up = (float*)d_ws;                       // [4096][32] f32
  float* wc = up + (size_t)Bc * Nc * Hp;          // [4096][32] f32 (1 MB total)

  dim3 g1(256, 4, 1);                              // 256 M-tiles x 4 N-tiles
  proj_wmma<<<g1, 32, 0, stream>>>(concepts, Ua, Wa, up, wc);

  dim3 g2(Nc / 16, Nc / 128, Bc);                  // (32, 4, 8)
  pair_score_wmma<<<g2, 256, 0, stream>>>(up, wc, va, out);
}
